// MolecularGCN_4105988735685
// MI455X (gfx1250) — matrix operand order repaired and checked by reading, not compiled, not run on hardware
//
#include <hip/hip_runtime.h>

#define N_NODES 100000
#define N_EDGES 1600000
#define F_IN 128
#define HID 64
#define N_GRAPHS 1024

typedef __attribute__((ext_vector_type(16))) _Float16 v16h;
typedef __attribute__((ext_vector_type(8)))  _Float16 v8h;
typedef __attribute__((ext_vector_type(4)))  _Float16 v4h;
typedef __attribute__((ext_vector_type(8)))  float    v8f;

// ---------------- degree / normalization ----------------
__global__ void k_deg(const int* __restrict__ dst, float* __restrict__ deg) {
  int t = blockIdx.x * blockDim.x + threadIdx.x;
  if (t < N_EDGES) atomicAdd(&deg[dst[t]], 1.0f);
}

__global__ void k_rsqrt(const float* __restrict__ deg, float* __restrict__ dis) {
  int t = blockIdx.x * blockDim.x + threadIdx.x;
  if (t < N_NODES) dis[t] = rsqrtf(deg[t] + 1.0f);
}

// ---------------- fp32 -> fp16 conversion (layer-1 input only) ----------------
__global__ void k_cvt_f16(const float* __restrict__ src, _Float16* __restrict__ dst, int n) {
  int t = blockIdx.x * blockDim.x + threadIdx.x;
  if (t < n) dst[t] = (_Float16)src[t];
}

// W[k][n] (KxHID row major) -> Wt[n][k] fp16 (K contiguous per n; each lane reads two 8-K runs)
__global__ void k_wt(const float* __restrict__ W, _Float16* __restrict__ Wt, int K) {
  int t = blockIdx.x * blockDim.x + threadIdx.x;
  if (t >= K * HID) return;
  int k = t / HID, n = t % HID;
  Wt[n * K + k] = (_Float16)W[k * HID + n];
}

// ---------------- WMMA GEMM + fused self-loop ----------------
// One wave computes a full 16 x HID row-strip (4 accumulators); A loaded once
// per K-step and reused across the 4 N-tiles.  Epilogue writes
//   XW  = A*B              (needed by the edge gather)
//   AGG = (A*B) * dis^2    (self-loop contribution, seed for the scatter)
// M % 16 == 0, K % 32 == 0.  Early-exit is wave-uniform so EXEC is all-1s.
__global__ void k_gemm_wmma(const _Float16* __restrict__ A, const _Float16* __restrict__ Bt,
                            const float* __restrict__ dis,
                            float* __restrict__ XW, float* __restrict__ AGG,
                            int M, int K) {
  int wave = (int)((blockIdx.x * blockDim.x + threadIdx.x) >> 5);
  int lane = threadIdx.x & 31;
  if (wave >= M / 16) return;
  int l16  = lane & 15;
  int half = lane >> 4;                       // 0: lanes 0-15, 1: lanes 16-31

  const _Float16* arow = A + (size_t)(wave * 16 + l16) * K;
  const _Float16* br0  = Bt + (size_t)(0 * 16 + l16) * K;
  const _Float16* br1  = Bt + (size_t)(1 * 16 + l16) * K;
  const _Float16* br2  = Bt + (size_t)(2 * 16 + l16) * K;
  const _Float16* br3  = Bt + (size_t)(3 * 16 + l16) * K;

  v8f acc0 = {}, acc1 = {}, acc2 = {}, acc3 = {};
  for (int k0 = 0; k0 < K; k0 += 32) {
    // 16-bit A 16x32 fragment: half 0 -> K=k0+[0..7],[16..23]; half 1 -> +8
    v8h a0 = *(const v8h*)(arow + k0 + half * 8);
    v8h a1 = *(const v8h*)(arow + k0 + half * 8 + 16);
    v16h a;
#pragma unroll
    for (int i = 0; i < 8; ++i) { a[i] = a0[i]; a[8 + i] = a1[i]; }

    // 16-bit B 32x16 fragment: half 0 -> K=k0+[0..7],[16..23]; half 1 -> +8
    const _Float16* brs[4] = {br0, br1, br2, br3};
    v8f* accs[4] = {&acc0, &acc1, &acc2, &acc3};
#pragma unroll
    for (int nt = 0; nt < 4; ++nt) {
      v8h b0 = *(const v8h*)(brs[nt] + k0 + half * 8);
      v8h b1 = *(const v8h*)(brs[nt] + k0 + half * 8 + 16);
      v16h b;
#pragma unroll
      for (int i = 0; i < 8; ++i) { b[i] = b0[i]; b[8 + i] = b1[i]; }
      *accs[nt] = __builtin_amdgcn_wmma_f32_16x16x32_f16(
          false, a, false, b, (short)0, *accs[nt], false, false);
    }
  }

  // C layout: VGPR i -> row (i + half*8), col l16 (+16*nt)
  float* xrow = XW  + (size_t)(wave * 16) * HID + l16;
  float* grow = AGG + (size_t)(wave * 16) * HID + l16;
  const v8f* accs[4] = {&acc0, &acc1, &acc2, &acc3};
#pragma unroll
  for (int i = 0; i < 8; ++i) {
    int r = i + half * 8;
    float s = dis[wave * 16 + r]; s *= s;
#pragma unroll
    for (int nt = 0; nt < 4; ++nt) {
      float v = (*accs[nt])[i];
      xrow[(size_t)r * HID + nt * 16] = v;
      grow[(size_t)r * HID + nt * 16] = v * s;
    }
  }
}

// ---------------- edge scatter: agg[dst] += xw[src] * dis[src]*dis[dst] ----------------
__global__ void k_edge(const int* __restrict__ src, const int* __restrict__ dst,
                       const float* __restrict__ dis, const float* __restrict__ xw,
                       float* __restrict__ agg) {
  long long t = (long long)blockIdx.x * blockDim.x + threadIdx.x;  // E*16 threads
  if (t >= (long long)N_EDGES * 16) return;
  int e = (int)(t >> 4);
  int c = ((int)t & 15) * 4;
  int s = src[e], d = dst[e];
  float w = dis[s] * dis[d];
  float4 v = *(const float4*)(xw + (size_t)s * HID + c);
  float* o = agg + (size_t)d * HID + c;
  atomicAdd(o + 0, v.x * w);
  atomicAdd(o + 1, v.y * w);
  atomicAdd(o + 2, v.z * w);
  atomicAdd(o + 3, v.w * w);
}

// ---------------- finalize: h = (relu?)(agg + bias), fused fp16 copy ----------------
__global__ void k_finalize(const float* __restrict__ agg, const float* __restrict__ bias,
                           float* __restrict__ h, _Float16* __restrict__ hf16, int do_relu) {
  int t = blockIdx.x * blockDim.x + threadIdx.x;           // N_NODES * 16 threads
  if (t >= N_NODES * 16) return;
  int c = (t & 15) * 4;
  float4 v = *(const float4*)(agg + (size_t)t * 4);
  float4 b = *(const float4*)(bias + c);
  float4 o;
  o.x = v.x + b.x; o.y = v.y + b.y; o.z = v.z + b.z; o.w = v.w + b.w;
  if (do_relu) {
    o.x = fmaxf(o.x, 0.0f); o.y = fmaxf(o.y, 0.0f);
    o.z = fmaxf(o.z, 0.0f); o.w = fmaxf(o.w, 0.0f);
  }
  *(float4*)(h + (size_t)t * 4) = o;
  if (hf16) {
    v4h p; p[0] = (_Float16)o.x; p[1] = (_Float16)o.y;
           p[2] = (_Float16)o.z; p[3] = (_Float16)o.w;
    *(v4h*)(hf16 + (size_t)t * 4) = p;
  }
}

// ---------------- pooling ----------------
__global__ void k_pool_cnt(const int* __restrict__ batch, float* __restrict__ cnt) {
  int t = blockIdx.x * blockDim.x + threadIdx.x;
  if (t < N_NODES) atomicAdd(&cnt[batch[t]], 1.0f);
}

__global__ void k_pool_sum(const float* __restrict__ h, const int* __restrict__ batch,
                           float* __restrict__ sums) {
  int t = blockIdx.x * blockDim.x + threadIdx.x;           // N_NODES * 16 threads
  if (t >= N_NODES * 16) return;
  int i = t >> 4, c = (t & 15) * 4;
  int g = batch[i];
  float4 v = *(const float4*)(h + (size_t)i * HID + c);
  float* o = sums + (size_t)g * HID + c;
  atomicAdd(o + 0, v.x);
  atomicAdd(o + 1, v.y);
  atomicAdd(o + 2, v.z);
  atomicAdd(o + 3, v.w);
}

__global__ void k_head(const float* __restrict__ sums, const float* __restrict__ cnt,
                       const float* __restrict__ Wl, const float* __restrict__ bl,
                       float* __restrict__ out) {
  int g = blockIdx.x * blockDim.x + threadIdx.x;
  if (g >= N_GRAPHS) return;
  float inv = 1.0f / fmaxf(cnt[g], 1.0f);
  float acc = bl[0];
#pragma unroll
  for (int f = 0; f < HID; ++f) acc += sums[(size_t)g * HID + f] * inv * Wl[f];
  out[g] = acc;
}

// ---------------- host orchestration ----------------
static inline int cdiv(long long a, int b) { return (int)((a + b - 1) / b); }

extern "C" void kernel_launch(void* const* d_in, const int* in_sizes, int n_in,
                              void* d_out, int out_size, void* d_ws, size_t ws_size,
                              hipStream_t stream) {
  const float* x     = (const float*)d_in[0];
  const int*   ei    = (const int*)d_in[1];       // [2, E]: src = ei, dst = ei + E
  const int*   batch = (const int*)d_in[2];
  const float* W1 = (const float*)d_in[3]; const float* b1 = (const float*)d_in[4];
  const float* W2 = (const float*)d_in[5]; const float* b2 = (const float*)d_in[6];
  const float* W3 = (const float*)d_in[7]; const float* b3 = (const float*)d_in[8];
  const float* Wl = (const float*)d_in[9]; const float* bl = (const float*)d_in[10];
  float* out = (float*)d_out;
  const int* src = ei;
  const int* dst = ei + N_EDGES;

  // workspace carve-up (256B aligned)
  char* p = (char*)d_ws;
  auto take = [&](size_t bytes) { char* r = p; p += (bytes + 255) & ~(size_t)255; return r; };
  float*    deg  = (float*)take((size_t)N_NODES * 4);
  float*    dis  = (float*)take((size_t)N_NODES * 4);
  float*    XW   = (float*)take((size_t)N_NODES * HID * 4);
  float*    AGG  = (float*)take((size_t)N_NODES * HID * 4);
  float*    H    = (float*)take((size_t)N_NODES * HID * 4);
  _Float16* HF16 = (_Float16*)take((size_t)N_NODES * F_IN * 2);
  _Float16* WT16 = (_Float16*)take((size_t)F_IN * HID * 2);
  float*    SUMS = (float*)take((size_t)N_GRAPHS * HID * 4);
  float*    CNT  = (float*)take((size_t)N_GRAPHS * 4);

  const int B = 256;
  const int gemm_threads = (N_NODES / 16) * 32;   // one wave per 16-row strip

  // ---- degree / normalization ----
  hipMemsetAsync(deg, 0, (size_t)N_NODES * 4, stream);
  k_deg<<<cdiv(N_EDGES, B), B, 0, stream>>>(dst, deg);
  k_rsqrt<<<cdiv(N_NODES, B), B, 0, stream>>>(deg, dis);

  // ---- layer 1: x [N,128] ----
  k_cvt_f16<<<cdiv((long long)N_NODES * F_IN, B), B, 0, stream>>>(x, HF16, N_NODES * F_IN);
  k_wt<<<cdiv(F_IN * HID, B), B, 0, stream>>>(W1, WT16, F_IN);
  k_gemm_wmma<<<cdiv(gemm_threads, B), B, 0, stream>>>(HF16, WT16, dis, XW, AGG, N_NODES, F_IN);
  k_edge<<<cdiv((long long)N_EDGES * 16, B), B, 0, stream>>>(src, dst, dis, XW, AGG);
  k_finalize<<<cdiv((long long)N_NODES * 16, B), B, 0, stream>>>(AGG, b1, H, HF16, 1);

  // ---- layers 2 and 3: H [N,64] (HF16 produced by previous finalize) ----
  const float* Ws[2] = {W2, W3};
  const float* bs[2] = {b2, b3};
  for (int L = 0; L < 2; ++L) {
    k_wt<<<cdiv(HID * HID, B), B, 0, stream>>>(Ws[L], WT16, HID);
    k_gemm_wmma<<<cdiv(gemm_threads, B), B, 0, stream>>>(HF16, WT16, dis, XW, AGG, N_NODES, HID);
    k_edge<<<cdiv((long long)N_EDGES * 16, B), B, 0, stream>>>(src, dst, dis, XW, AGG);
    k_finalize<<<cdiv((long long)N_NODES * 16, B), B, 0, stream>>>(
        AGG, bs[L], H, (L == 0) ? HF16 : (_Float16*)nullptr, (L == 0) ? 1 : 0);
  }

  // ---- pooling + head ----
  hipMemsetAsync(SUMS, 0, (size_t)N_GRAPHS * HID * 4, stream);
  hipMemsetAsync(CNT, 0, (size_t)N_GRAPHS * 4, stream);
  k_pool_cnt<<<cdiv(N_NODES, B), B, 0, stream>>>(batch, CNT);
  k_pool_sum<<<cdiv((long long)N_NODES * 16, B), B, 0, stream>>>(H, batch, SUMS);
  k_head<<<cdiv(N_GRAPHS, B), B, 0, stream>>>(SUMS, CNT, Wl, bl, out);
}
